// RNN_15796889715173
// MI455X (gfx1250) — matrix operand order repaired and checked by reading, not compiled, not run on hardware
//
#include <hip/hip_runtime.h>
#include <stdint.h>

// ---------------- types ----------------
typedef __attribute__((ext_vector_type(16))) __bf16 v16bf;
typedef __attribute__((ext_vector_type(8)))  __bf16 v8bf;
typedef __attribute__((ext_vector_type(8)))  float  v8f;

// explicitly global-address-space pointers (keeps loads on the GLOBAL path, not FLAT)
typedef const __attribute__((address_space(1))) unsigned short* gptr_u16;
typedef const __attribute__((address_space(1))) v8bf*           gptr_v8bf;

// ---------------- dims ----------------
#define BATCH 256
#define SEQ   80
#define EMBD  100
#define KX0   128        // EMBD padded up to multiple of 32
#define U     1024
#define G4    4096       // 4*U (gates i,f,g,o)
#define NCLS  2

// ---------------- CDNA5 async-to-LDS (inline asm) ----------------
// GLOBAL_LOAD_ASYNC_TO_LDS_B128: 16-byte global->LDS copy, ASYNCcnt-tracked, bypasses VGPRs.
__device__ __forceinline__ void cp16_g2l(const unsigned short* g, unsigned short* l) {
#if defined(__gfx1250__)
    unsigned lds_off = (unsigned)(uintptr_t)(__attribute__((address_space(3))) unsigned short*)l;
    unsigned long long gaddr = (unsigned long long)(uintptr_t)g;
    asm volatile("global_load_async_to_lds_b128 %0, %1, off"
                 :
                 : "v"(lds_off), "v"(gaddr)
                 : "memory");
#else
    *(v8bf*)l = *(const v8bf*)g;     // non-gfx1250 fallback
#endif
}
__device__ __forceinline__ void wait_async_le4() {
#if defined(__gfx1250__)
    asm volatile("s_wait_asynccnt 4" ::: "memory");
#endif
}
__device__ __forceinline__ void wait_async_0() {
#if defined(__gfx1250__)
    asm volatile("s_wait_asynccnt 0" ::: "memory");
#endif
}

// ---------------- scalar helpers ----------------
__device__ __forceinline__ unsigned short f2bf(float f) {
    unsigned int u = __float_as_uint(f);
    u += 0x7fffu + ((u >> 16) & 1u);          // round-to-nearest-even
    return (unsigned short)(u >> 16);
}
__device__ __forceinline__ float bf2f(unsigned short h) {
    return __uint_as_float(((unsigned int)h) << 16);
}
__device__ __forceinline__ float sigmoidf(float x) {
    return 1.0f / (1.0f + __expf(-x));
}

// A fragment: 16x32 bf16 tile from row-major A[row][k] (rows m0..m0+15, k0..k0+31).
// ISA layout (16-bit A 16x32): lanes 0-15 hold K {0..7,16..23}, lanes 16-31 hold {8..15,24..31}.
__device__ __forceinline__ v16bf load_a(gptr_u16 A, int strideA, int m0, int k0, int lane) {
    int row = m0 + (lane & 15);
    int klo = k0 + ((lane >> 4) << 3);                 // +0 lanes<16, +8 else
    gptr_u16 p = A + (size_t)row * strideA + klo;
    v8bf lo = *(gptr_v8bf)p;                           // K = klo    .. klo+7
    v8bf hi = *(gptr_v8bf)(p + 16);                    // K = klo+16 .. klo+23
    return __builtin_shufflevector(lo, hi, 0,1,2,3,4,5,6,7,8,9,10,11,12,13,14,15);
}

// B fragment from LDS tile ldsB[col][32] (col = n-in-tile): lanes 0-15 K {0..7,16..23}, lanes 16-31 K {8..15,24..31}.
__device__ __forceinline__ v16bf load_b_lds(const unsigned short* base, int lane) {
    const unsigned short* p = base + (lane & 15) * 32 + ((lane >> 4) << 3);
    v8bf lo = *(const v8bf*)p;                         // K = klo .. klo+7
    v8bf hi = *(const v8bf*)(p + 16);                  // K = klo+16.. klo+23
    return __builtin_shufflevector(lo, hi, 0,1,2,3,4,5,6,7,8,9,10,11,12,13,14,15);
}

// ---------------- weight convert + transpose: W[K][N] f32 -> WT[N][Kpad] bf16 ----------------
__global__ void convert_wT(const float* __restrict__ W, unsigned short* __restrict__ WT,
                           int K, int Kpad, int N) {
    int idx = blockIdx.x * blockDim.x + threadIdx.x;
    if (idx >= N * Kpad) return;
    int n = idx / Kpad, k = idx - n * Kpad;
    float v = (k < K) ? W[(size_t)k * N + n] : 0.0f;
    WT[idx] = f2bf(v);
}

// ---------------- embedding gather for all timesteps ----------------
__global__ void embed_all(const int* __restrict__ tokens, const float* __restrict__ emb,
                          unsigned short* __restrict__ x_all) {
    int idx = blockIdx.x * blockDim.x + threadIdx.x;   // BATCH*SEQ*KX0 threads
    int d  = idx & (KX0 - 1);
    int bt = idx >> 7;
    int tok = tokens[bt];
    float v = (d < EMBD) ? emb[(size_t)tok * EMBD + d] : 0.0f;
    x_all[idx] = f2bf(v);
}

__global__ void zero_u32(unsigned int* __restrict__ p, int n) {
    int i = blockIdx.x * blockDim.x + threadIdx.x;
    if (i < n) p[i] = 0u;
}

// ---------------- fused LSTM step ----------------
// Block tile: M=64 batch rows x N=64 units (x 4 gates). 8 waves:
//   mt = wave>>2 selects 32-row half (2 x 16-row WMMA subtiles per wave),
//   nt = wave&3 selects 16 unit-cols. 8 WMMAs per wave per k-iter.
// B tiles (64 cols x 4 gates x 32 k = 16 KB) streamed global->LDS with
// ASYNCcnt-tracked async copies, double buffered; shared by both mt groups.
__global__ __launch_bounds__(256)
void lstm_step(const unsigned short* __restrict__ Ax, int strideAx, int Kx,
               const unsigned short* __restrict__ WxT,          // [G4][Kx] bf16
               const unsigned short* __restrict__ Ah,           // [BATCH][U] bf16
               const unsigned short* __restrict__ WhT,          // [G4][U] bf16
               const float* __restrict__ bias,                  // [G4]
               float* __restrict__ c,                           // [BATCH][U] in-place
               unsigned short* __restrict__ hout)               // [BATCH][U] bf16
{
    __shared__ unsigned short ldsB[2][256][32];                 // 2 x 16 KB

    const int tid  = threadIdx.x;
    const int lane = tid & 31;
    const int wave = tid >> 5;
    const int r0 = blockIdx.x * 64 + (wave >> 2) * 32;          // this wave's 32 batch rows
    const int nt = wave & 3;                                    // 16 unit-cols within block tile

    // copy duty: thread tid fills LDS row tid = (gate = tid>>6, ncol = tid&63)
    const int copyRow = blockIdx.y * 64 + (tid & 63) + (tid >> 6) * U;   // WT row (0..4095)
    const unsigned short* src0 = WxT + (size_t)copyRow * Kx;    // phase-0 copy base
    const unsigned short* src1 = WhT + (size_t)copyRow * U;     // phase-1 copy base

    // keep A pointers in the GLOBAL address space across the phase select
    const gptr_u16 AxG = (gptr_u16)Ax;
    const gptr_u16 AhG = (gptr_u16)Ah;

    const int nIt0 = Kx / 32;
    const int nTot = nIt0 + U / 32;

    v8f acc[2][4] = {};                                         // [m-subtile][gate]

    // prologue: stage k-iter 0 into buffer 0
    #pragma unroll
    for (int j = 0; j < 4; ++j) cp16_g2l(src0 + j * 8, &ldsB[0][tid][j * 8]);

    int buf = 0;
    for (int it = 0; it < nTot; ++it) {
        const bool have_nxt = (it + 1) < nTot;
        if (have_nxt) {                                         // prefetch next tile
            int nx  = it + 1;
            bool ph1 = (nx >= nIt0);
            int  k0n = (ph1 ? (nx - nIt0) : nx) * 32;
            const unsigned short* src = (ph1 ? src1 : src0) + k0n;
            #pragma unroll
            for (int j = 0; j < 4; ++j) cp16_g2l(src + j * 8, &ldsB[buf ^ 1][tid][j * 8]);
            wait_async_le4();                                   // current tile's copies done
        } else {
            wait_async_0();
        }
        __syncthreads();                                        // whole block's tile visible

        const bool ph1 = (it >= nIt0);
        const int  k0  = (ph1 ? (it - nIt0) : it) * 32;
        const gptr_u16 Ap = ph1 ? AhG : AxG;
        const int Astride = ph1 ? U : strideAx;
        v16bf a0 = load_a(Ap, Astride, r0,      k0, lane);
        v16bf a1 = load_a(Ap, Astride, r0 + 16, k0, lane);
        #pragma unroll
        for (int g = 0; g < 4; ++g) {
            v16bf b = load_b_lds(&ldsB[buf][g * 64 + nt * 16][0], lane);
            acc[0][g] = __builtin_amdgcn_wmma_f32_16x16x32_bf16(
                false, a0, false, b, (short)0, acc[0][g], false, false);
            acc[1][g] = __builtin_amdgcn_wmma_f32_16x16x32_bf16(
                false, a1, false, b, (short)0, acc[1][g], false, false);
        }
        __syncthreads();                                        // done reading before overwrite
        buf ^= 1;
    }

    // Epilogue. C/D layout: (VGPR r, lane L): row = base + r + (L>=16 ? 8 : 0), col = u0 + (L&15)
    const int col = blockIdx.y * 64 + nt * 16 + (lane & 15);
    const float bi = bias[col];
    const float bf = bias[col + U];
    const float bg = bias[col + 2 * U];
    const float bo = bias[col + 3 * U];
    const int rofs = (lane >> 4) << 3;

    #pragma unroll
    for (int m = 0; m < 2; ++m) {
        #pragma unroll
        for (int vr = 0; vr < 8; ++vr) {
            int row = r0 + m * 16 + vr + rofs;
            size_t idx = (size_t)row * U + col;
            float zi = acc[m][0][vr] + bi;
            float zf = acc[m][1][vr] + bf;
            float zg = acc[m][2][vr] + bg;
            float zo = acc[m][3][vr] + bo;
            float cn = sigmoidf(zf) * c[idx] + sigmoidf(zi) * tanhf(zg);
            float hn = sigmoidf(zo) * tanhf(cn);
            c[idx] = cn;
            hout[idx] = f2bf(hn);
        }
    }
}

// ---------------- per-step Dense head ----------------
__global__ void fc_step(const unsigned short* __restrict__ h1,
                        const float* __restrict__ fcW, const float* __restrict__ fcb,
                        float* __restrict__ out /* pre-offset by t*NCLS */) {
    int tid = blockIdx.x * blockDim.x + threadIdx.x;   // 0..511
    int b = tid >> 1, cls = tid & 1;
    const unsigned short* hr = h1 + (size_t)b * U;
    float s = fcb[cls];
    for (int u = 0; u < U; ++u)
        s += bf2f(hr[u]) * fcW[u * NCLS + cls];
    out[(size_t)b * (SEQ * NCLS) + cls] = s;
}

// ---------------- host launch ----------------
extern "C" void kernel_launch(void* const* d_in, const int* in_sizes, int n_in,
                              void* d_out, int out_size, void* d_ws, size_t ws_size,
                              hipStream_t stream) {
    (void)in_sizes; (void)n_in; (void)out_size; (void)ws_size;
    const int*   tokens = (const int*)  d_in[0];
    const float* emb    = (const float*)d_in[1];
    const float* W0x    = (const float*)d_in[2];
    const float* W0h    = (const float*)d_in[3];
    const float* b0     = (const float*)d_in[4];
    const float* W1x    = (const float*)d_in[5];
    const float* W1h    = (const float*)d_in[6];
    const float* b1     = (const float*)d_in[7];
    const float* fcW    = (const float*)d_in[8];
    const float* fcb    = (const float*)d_in[9];
    float* out = (float*)d_out;

    // workspace carve-out (~36 MB)
    uint8_t* p = (uint8_t*)d_ws;
    auto take = [&](size_t bytes) {
        uint8_t* r = p; p += (bytes + 255) & ~(size_t)255; return r;
    };
    unsigned short* W0xT  = (unsigned short*)take((size_t)G4 * KX0 * 2);
    unsigned short* W0hT  = (unsigned short*)take((size_t)G4 * U * 2);
    unsigned short* W1xT  = (unsigned short*)take((size_t)G4 * U * 2);
    unsigned short* W1hT  = (unsigned short*)take((size_t)G4 * U * 2);
    unsigned short* xall  = (unsigned short*)take((size_t)BATCH * SEQ * KX0 * 2);
    unsigned short* h0b[2], *h1b[2];
    h0b[0] = (unsigned short*)take((size_t)BATCH * U * 2);
    h0b[1] = (unsigned short*)take((size_t)BATCH * U * 2);
    h1b[0] = (unsigned short*)take((size_t)BATCH * U * 2);
    h1b[1] = (unsigned short*)take((size_t)BATCH * U * 2);
    float* c0 = (float*)take((size_t)BATCH * U * 4);
    float* c1 = (float*)take((size_t)BATCH * U * 4);

    // one-time prep
    convert_wT<<<(G4 * KX0) / 256, 256, 0, stream>>>(W0x, W0xT, EMBD, KX0, G4);
    convert_wT<<<(G4 * U) / 256, 256, 0, stream>>>(W0h, W0hT, U, U, G4);
    convert_wT<<<(G4 * U) / 256, 256, 0, stream>>>(W1x, W1xT, U, U, G4);
    convert_wT<<<(G4 * U) / 256, 256, 0, stream>>>(W1h, W1hT, U, U, G4);
    embed_all<<<(BATCH * SEQ * KX0) / 256, 256, 0, stream>>>(tokens, emb, xall);
    zero_u32<<<(BATCH * U / 2) / 256, 256, 0, stream>>>((unsigned int*)h0b[0], BATCH * U / 2);
    zero_u32<<<(BATCH * U / 2) / 256, 256, 0, stream>>>((unsigned int*)h1b[0], BATCH * U / 2);
    zero_u32<<<(BATCH * U) / 256, 256, 0, stream>>>((unsigned int*)c0, BATCH * U);
    zero_u32<<<(BATCH * U) / 256, 256, 0, stream>>>((unsigned int*)c1, BATCH * U);

    dim3 grid(BATCH / 64, U / 64);   // (4,16) blocks of 8 waves
    for (int t = 0; t < SEQ; ++t) {
        const unsigned short* h0r = h0b[t & 1];
        unsigned short*       h0w = h0b[(t + 1) & 1];
        const unsigned short* h1r = h1b[t & 1];
        unsigned short*       h1w = h1b[(t + 1) & 1];

        // layer 0: x_t (K=128 padded) + h0 (K=1024)
        lstm_step<<<grid, 256, 0, stream>>>(xall + (size_t)t * KX0, SEQ * KX0, KX0,
                                            W0xT, h0r, W0hT, b0, c0, h0w);
        // layer 1: h0_t (K=1024) + h1 (K=1024)
        lstm_step<<<grid, 256, 0, stream>>>(h0w, U, U,
                                            W1xT, h1r, W1hT, b1, c1, h1w);
        // dense head for this timestep
        fc_step<<<2, 256, 0, stream>>>(h1w, fcW, fcb, out + t * NCLS);
    }
}
